// RNNRBM_64613488001651
// MI455X (gfx1250) — matrix operand order repaired and checked by reading, not compiled, not run on hardware
//
#include <hip/hip_runtime.h>
#include <hip/hip_bf16.h>
#include <stdint.h>

typedef __attribute__((ext_vector_type(16))) _Float16 v16h;
typedef __attribute__((ext_vector_type(8)))  _Float16 v8h;
typedef __attribute__((ext_vector_type(8)))  float    v8f;

#define V_DIM  88
#define VP     96      // V padded to multiple of 32 for WMMA K / 16 for tiles
#define H_DIM  256
#define R_DIM  256
#define T_DIM  4096
#define NT     4095    // scan length
#define GIBBS  20

// ---------------- workspace (device scratch) layout, all f16 64B-aligned ----
#define WS_WUH   0
#define WS_WUV   (WS_WUH + 256*256)
#define WS_WVU   (WS_WUV + 96*256)
#define WS_WUU   (WS_WVU + 256*96)
#define WS_WA    (WS_WUU + 256*256)
#define WS_WT    (WS_WA  + 256*96)
#define WS_HALVES (WS_WT + 96*256)     // = 229376 halves = 448KB

// ---------------- dynamic LDS layout (bytes, all offsets 64B aligned) -------
#define OFF_WA    0                    // _Float16[256*96]  48KB
#define OFF_WT    49152                // _Float16[96*256]  48KB
#define OFF_UH    98304                // _Float16[256]
#define OFF_VH    98816                // _Float16[128]
#define OFF_HH    99072                // _Float16[256]
#define OFF_VTH   99584                // _Float16[128]
#define OFF_BH    99840                // float[256]
#define OFF_BV    100864               // float[128]
#define OFF_TMP   101376               // float[256]
#define OFF_LOSS  102400               // float[128]
#define OFF_MSE   102912               // float[128]
#define OFF_VTF   103424               // float[128]
#define SMEM_TOTAL 103936

// ---------------- helpers ---------------------------------------------------
__device__ __forceinline__ float rnd01(uint32_t a, uint32_t b, uint32_t c) {
  uint32_t x = a * 0x9E3779B9u + b * 0x85EBCA6Bu + c * 0xC2B2AE35u + 0x27220A95u;
  x ^= x >> 16; x *= 0x7FEB352Du; x ^= x >> 15; x *= 0x846CA68Bu; x ^= x >> 16;
  return (float)(x >> 8) * (1.0f / 16777216.0f);
}

__device__ __forceinline__ float sigmoidf(float x) { return 1.0f / (1.0f + __expf(-x)); }

// A fragment, 16x32 f16 tile at (row0, k0) of row-major matrix with given stride.
// lanes 0-15 -> M=lane, K = k0..k0+15;
// lanes 16-31 -> M=lane-16, K = k0+16..k0+31 (matches B contiguous order).
__device__ __forceinline__ v16h load_a_frag(const _Float16* mat, int stride,
                                            int row0, int k0, int lane) {
  const int m    = lane & 15;
  const int koff = (lane & 16) ? 16 : 0;
  const _Float16* p = mat + (row0 + m) * stride + k0 + koff;
  v8h lo = *(const v8h*)(p);        // 16B aligned (stride mult of 8 halves)
  v8h hi = *(const v8h*)(p + 8);
  return __builtin_shufflevector(lo, hi, 0,1,2,3,4,5,6,7,8,9,10,11,12,13,14,15);
}

// B fragments for a mat-vec, preloaded once per phase and reused across tiles.
// ALL lanes load the chunk their half-group owns (broadcast LDS read):
// lanes 0-15 carry K=k0..k0+15 in their column, lanes 16-31 K=k0+16..k0+31.
// Every B column then equals the vector, so every D column equals the result;
// we simply read column 0 from lanes 0/16. No exec masking, no zeroing movs.
template<int K>
__device__ __forceinline__ void load_b_all(const _Float16* vec, int lane, v16h* b) {
#pragma unroll
  for (int i = 0; i < K / 32; ++i)
    b[i] = *(const v16h*)(vec + i * 32 + (lane & 16));   // 32B aligned
}

template<int K>
__device__ __forceinline__ v8f wmma_matvec_acc(const _Float16* mat, int stride,
                                               int row0, const v16h* b,
                                               int lane, v8f c) {
#pragma unroll
  for (int k0 = 0; k0 < K; k0 += 32) {
    v16h a = load_a_frag(mat, stride, row0, k0, lane);
    c = __builtin_amdgcn_wmma_f32_16x16x32_f16(
        /*neg_a=*/false, a, /*neg_b=*/false, b[k0 / 32],
        /*c_mod=*/(short)0, c, /*reuse_a=*/false, /*reuse_b=*/false);
  }
  return c;
}

// D column 0 lives in lanes 0 (M=row0..row0+7) and 16 (M=row0+8..row0+15).
__device__ __forceinline__ void store_col0(v8f c, float* out, int row0, int lane) {
  if (lane == 0) {
#pragma unroll
    for (int r = 0; r < 8; ++r) out[row0 + r] = c[r];
  } else if (lane == 16) {
#pragma unroll
    for (int r = 0; r < 8; ++r) out[row0 + 8 + r] = c[r];
  }
}

// ---------------- kernel 1: pack weights to padded f16 ----------------------
__global__ void pack_weights(const float* __restrict__ w,   const float* __restrict__ wuu,
                             const float* __restrict__ wuv, const float* __restrict__ wuh,
                             const float* __restrict__ wvu, _Float16* __restrict__ ws) {
  const int tid  = blockIdx.x * blockDim.x + threadIdx.x;
  const int nthr = gridDim.x * blockDim.x;
  for (int i = tid; i < 256 * 256; i += nthr) {
    ws[WS_WUH + i] = (_Float16)wuh[i];
    ws[WS_WUU + i] = (_Float16)wuu[i];
  }
  for (int i = tid; i < 96 * 256; i += nthr) {            // rows padded 88->96
    int r = i >> 8, c = i & 255;
    ws[WS_WUV + i] = (r < V_DIM) ? (_Float16)wuv[r * 256 + c] : (_Float16)0.f;
    ws[WS_WT  + i] = (r < V_DIM) ? (_Float16)w[c * V_DIM + r] : (_Float16)0.f; // w^T
  }
  for (int i = tid; i < 256 * 96; i += nthr) {            // cols padded 88->96
    int r = i / 96, c = i % 96;
    ws[WS_WVU + i] = (c < V_DIM) ? (_Float16)wvu[r * V_DIM + c] : (_Float16)0.f;
    ws[WS_WA  + i] = (c < V_DIM) ? (_Float16)w[r * V_DIM + c]   : (_Float16)0.f;
  }
}

// ---------------- kernel 2: reg term = 0.2*(||wuv||_F + ||wuh||_F) ----------
__global__ void reg_kernel(const float* __restrict__ wuv, const float* __restrict__ wuh,
                           float* __restrict__ regp, float* __restrict__ out) {
  __shared__ float s1[256], s2[256];
  float a = 0.f, b = 0.f;
  for (int i = threadIdx.x; i < V_DIM * 256; i += 256) a += wuv[i] * wuv[i];
  for (int i = threadIdx.x; i < 256 * 256; i += 256)   b += wuh[i] * wuh[i];
  s1[threadIdx.x] = a; s2[threadIdx.x] = b;
  __syncthreads();
  for (int st = 128; st > 0; st >>= 1) {
    if (threadIdx.x < st) { s1[threadIdx.x] += s1[threadIdx.x + st];
                            s2[threadIdx.x] += s2[threadIdx.x + st]; }
    __syncthreads();
  }
  if (threadIdx.x == 0) {
    float reg = 0.2f * (sqrtf(s1[0]) + sqrtf(s2[0]));
    regp[0] = reg;
    out[1 + NT] = reg;                 // out[4096] = reg_term
  }
}

// ---------------- kernel 3: persistent sequential scan -----------------------
__global__ void __launch_bounds__(256, 1)
rnnrbm_scan(const float* __restrict__ visible,
            const float* __restrict__ bv, const float* __restrict__ bh,
            const float* __restrict__ bu, const float* __restrict__ u0,
            const _Float16* __restrict__ ws, const float* __restrict__ regp,
            float* __restrict__ out) {
  extern __shared__ char smem[];
  _Float16* sWA  = (_Float16*)(smem + OFF_WA);    // w   [256][96]
  _Float16* sWT  = (_Float16*)(smem + OFF_WT);    // w^T [96][256]
  _Float16* sUh  = (_Float16*)(smem + OFF_UH);    // u_{t-1} (f16)
  _Float16* sVh  = (_Float16*)(smem + OFF_VH);    // gibbs visible state (f16, padded)
  _Float16* sHh  = (_Float16*)(smem + OFF_HH);    // sampled hidden (f16)
  _Float16* sVth = (_Float16*)(smem + OFF_VTH);   // v_t (f16, padded)
  float* sBh   = (float*)(smem + OFF_BH);
  float* sBv   = (float*)(smem + OFF_BV);
  float* sTmp  = (float*)(smem + OFF_TMP);
  float* sLoss = (float*)(smem + OFF_LOSS);
  float* sMse  = (float*)(smem + OFF_MSE);
  float* sVtf  = (float*)(smem + OFF_VTF);        // v_t (f32) for loss

  const _Float16* wuhA = ws + WS_WUH;
  const _Float16* wuvA = ws + WS_WUV;
  const _Float16* wvuA = ws + WS_WVU;
  const _Float16* wuuA = ws + WS_WUU;

  const int tid  = threadIdx.x;
  const int lane = tid & 31;
  const int wid  = tid >> 5;

  // Stage hottest matrices (w, w^T) into LDS: 96KB, used 40 matvecs/timestep.
  {
    const uint4* srcA = (const uint4*)(ws + WS_WA);
    const uint4* srcT = (const uint4*)(ws + WS_WT);
    uint4* dstA = (uint4*)sWA;
    uint4* dstT = (uint4*)sWT;
    for (int i = tid; i < (256 * 96 * 2) / 16; i += 256) { dstA[i] = srcA[i]; dstT[i] = srcT[i]; }
  }
  sUh[tid] = (_Float16)u0[tid];
  __syncthreads();

  float total = 0.f;   // only thread 0's copy is meaningful

  for (int t = 0; t < NT; ++t) {
    // ---- frame load: v_{t-1} seeds the Gibbs chain, v_t for loss/RNN -------
    if (tid < VP) {
      float vprev = (tid < V_DIM) ? visible[t * V_DIM + tid]       : 0.f;
      float vcur  = (tid < V_DIM) ? visible[(t + 1) * V_DIM + tid] : 0.f;
      sVh[tid]  = (_Float16)vprev;
      sVth[tid] = (_Float16)vcur;
      sVtf[tid] = vcur;
    }
    __syncthreads();

    // ---- dynamic biases: bh_t = wuh@u, bv_t = wuv@u (raw); shared B = u ----
    {
      v16h bu8[8];
      load_b_all<256>(sUh, lane, bu8);
#pragma unroll
      for (int tile = wid; tile < 16; tile += 8) {
        v8f c = {};
        c = wmma_matvec_acc<256>(wuhA, 256, tile * 16, bu8, lane, c);
        store_col0(c, sBh, tile * 16, lane);
      }
      if (wid < 6) {
        v8f c = {};
        c = wmma_matvec_acc<256>(wuvA, 256, wid * 16, bu8, lane, c);
        store_col0(c, sBv, wid * 16, lane);
      }
    }
    __syncthreads();
    sBh[tid] += bh[tid];
    if (tid < V_DIM) sBv[tid] += bv[tid];
    __syncthreads();

    // ---- Gibbs chain (20 dependent CD steps) -------------------------------
    for (int g = 0; g < GIBBS; ++g) {
      // h_probs = sigmoid(w @ v + bh_t); sample
      {
        v16h bv3[3];
        load_b_all<VP>(sVh, lane, bv3);
#pragma unroll
        for (int tile = wid; tile < 16; tile += 8) {
          v8f c = {};
          c = wmma_matvec_acc<VP>(sWA, VP, tile * 16, bv3, lane, c);
          store_col0(c, sTmp, tile * 16, lane);
        }
      }
      __syncthreads();
      {
        float p = sigmoidf(sTmp[tid] + sBh[tid]);
        sHh[tid] = (_Float16)((p > rnd01((uint32_t)t, (uint32_t)(2 * g), (uint32_t)tid)) ? 1.f : 0.f);
      }
      __syncthreads();
      // v_probs = sigmoid(w^T @ h + bv_t); sample
      {
        v16h bh8[8];
        load_b_all<256>(sHh, lane, bh8);
        if (wid < 6) {
          v8f c = {};
          c = wmma_matvec_acc<256>(sWT, 256, wid * 16, bh8, lane, c);
          store_col0(c, sTmp, wid * 16, lane);
        }
      }
      __syncthreads();
      if (tid < VP) {
        float smp = 0.f;
        if (tid < V_DIM) {
          float p = sigmoidf(sTmp[tid] + sBv[tid]);
          smp = (p > rnd01((uint32_t)t, (uint32_t)(2 * g + 1), (uint32_t)tid)) ? 1.f : 0.f;
        }
        sVh[tid] = (_Float16)smp;
      }
      __syncthreads();
    }

    // ---- cross-entropy vs y_t = sigmoid(bv_t), and mse vs negative sample --
    {
      float l = 0.f, m = 0.f;
      if (tid < V_DIM) {
        float vt = sVtf[tid];
        float y  = sigmoidf(sBv[tid]);
        l = -vt * __logf(1e-6f + y) - (1.f - vt) * __logf(1e-6f + 1.f - y);
        m = fabsf(vt - (float)sVh[tid]);
      }
      // wave32 shuffle-tree reduction; threads 0..87 live in waves 0..2
#pragma unroll
      for (int off = 16; off > 0; off >>= 1) {
        l += __shfl_down(l, off);
        m += __shfl_down(m, off);
      }
      if (lane == 0 && wid < 3) { sLoss[wid] = l; sMse[wid] = m; }
    }
    __syncthreads();
    if (tid == 0) {
      total += sLoss[0] + sLoss[1] + sLoss[2];
      out[1 + t] = (sMse[0] + sMse[1] + sMse[2]) * (1.f / (float)V_DIM);
    }

    // ---- RNN update: u_t = tanh(wvu@v_t + wuu@u_{t-1} + bu) ----------------
    {
      v16h bvt3[3];
      v16h buu8[8];
      load_b_all<VP>(sVth, lane, bvt3);
      load_b_all<256>(sUh, lane, buu8);
#pragma unroll
      for (int tile = wid; tile < 16; tile += 8) {
        v8f c = {};
        c = wmma_matvec_acc<VP>(wvuA, VP, tile * 16, bvt3, lane, c);
        c = wmma_matvec_acc<256>(wuuA, 256, tile * 16, buu8, lane, c);
        store_col0(c, sTmp, tile * 16, lane);
      }
    }
    __syncthreads();                         // all reads of old sUh complete
    {
      float un = tanhf(sTmp[tid] + bu[tid]);
      sUh[tid] = (_Float16)un;
    }
    __syncthreads();
  }

  if (tid == 0) out[0] = total / (float)T_DIM + regp[0];
}

// ---------------- launch -----------------------------------------------------
extern "C" void kernel_launch(void* const* d_in, const int* in_sizes, int n_in,
                              void* d_out, int out_size, void* d_ws, size_t ws_size,
                              hipStream_t stream) {
  const float* visible = (const float*)d_in[0];
  const float* w   = (const float*)d_in[1];
  const float* wuu = (const float*)d_in[2];
  const float* wuv = (const float*)d_in[3];
  const float* wuh = (const float*)d_in[4];
  const float* wvu = (const float*)d_in[5];
  const float* bv  = (const float*)d_in[6];
  const float* bh  = (const float*)d_in[7];
  const float* bu  = (const float*)d_in[8];
  const float* u0  = (const float*)d_in[9];
  float* out = (float*)d_out;

  _Float16* wsh = (_Float16*)d_ws;
  float* regp = (float*)((char*)d_ws + (size_t)WS_HALVES * sizeof(_Float16)); // 64B aligned

  pack_weights<<<64, 256, 0, stream>>>(w, wuu, wuv, wuh, wvu, wsh);
  reg_kernel<<<1, 256, 0, stream>>>(wuv, wuh, regp, out);
  rnnrbm_scan<<<1, 256, SMEM_TOTAL, stream>>>(visible, bv, bh, bu, u0, wsh, regp, out);
}
